// LSTM_72576357367871
// MI455X (gfx1250) — matrix operand order repaired and checked by reading, not compiled, not run on hardware
//
#include <hip/hip_runtime.h>
#include <hip/hip_bf16.h>

// LSTM on gfx1250: fused [x_t, h] @ [[W_in],[W_h]] per step via v_wmma_f32_16x16x32_f16.
// x is pre-converted/transposed to f16 [T,B,I] once; weights packed N-major f16 (L2-resident).
// Each wave owns a 16x16 tile across all 4 gates -> wave-local pointwise c/h update.

typedef _Float16 h16_t;
typedef __attribute__((ext_vector_type(16))) _Float16 v16h;
typedef __attribute__((ext_vector_type(8)))  float    v8f;

#define B_    128
#define T_    512
#define I_    1024
#define H_    1024
#define KTOT  2048   // I + H

// ---------------------------------------------------------------------------
// Prep: pack W into Wt[4H][KTOT] f16, Wt[g*H+n][k] = (k<I ? Wi_g[k][n] : Wh_g[k-I][n])
// ---------------------------------------------------------------------------
__global__ void pack_weights(const float* __restrict__ Wii, const float* __restrict__ Wif,
                             const float* __restrict__ Wig, const float* __restrict__ Wio,
                             const float* __restrict__ Whi, const float* __restrict__ Whf,
                             const float* __restrict__ Whg, const float* __restrict__ Who,
                             h16_t* __restrict__ Wt) {
    int idx = blockIdx.x * blockDim.x + threadIdx.x;   // over 4096*2048
    int k   = idx & (KTOT - 1);
    int row = idx >> 11;              // 0..4095
    int g   = row >> 10;              // gate 0..3 (i,f,g,o)
    int n   = row & (H_ - 1);
    const float* Wi[4] = {Wii, Wif, Wig, Wio};
    const float* Wh[4] = {Whi, Whf, Whg, Who};
    float v = (k < I_) ? Wi[g][(size_t)k * H_ + n]
                       : Wh[g][(size_t)(k - I_) * H_ + n];
    Wt[idx] = (h16_t)v;
}

// ---------------------------------------------------------------------------
// Prep: x [B,T,I] fp32 -> xh [T,B,I] f16 (8 elements per thread, vector io)
// ---------------------------------------------------------------------------
__global__ void convert_x(const float* __restrict__ x, h16_t* __restrict__ xh) {
    size_t e = ((size_t)blockIdx.x * blockDim.x + threadIdx.x) * 8;  // < 64M
    int i = (int)(e & (I_ - 1));
    int t = (int)((e >> 10) & (T_ - 1));
    int b = (int)(e >> 19);
    float4 a0 = ((const float4*)(x + e))[0];
    float4 a1 = ((const float4*)(x + e))[1];
    h16_t tmp[8];
    tmp[0]=(h16_t)a0.x; tmp[1]=(h16_t)a0.y; tmp[2]=(h16_t)a0.z; tmp[3]=(h16_t)a0.w;
    tmp[4]=(h16_t)a1.x; tmp[5]=(h16_t)a1.y; tmp[6]=(h16_t)a1.z; tmp[7]=(h16_t)a1.w;
    *(uint4*)(xh + ((size_t)t * B_ + b) * I_ + i) = *(const uint4*)tmp;
}

// ---------------------------------------------------------------------------
// Prep: combined bias (b_i* + b_h*) and zero-init of h0 (f16) and c (fp32).
// ---------------------------------------------------------------------------
__global__ void prep_bias_init(const float* __restrict__ bii, const float* __restrict__ bhi,
                               const float* __restrict__ bif, const float* __restrict__ bhf,
                               const float* __restrict__ big, const float* __restrict__ bhg,
                               const float* __restrict__ bio, const float* __restrict__ bho,
                               float* __restrict__ bcomb, h16_t* __restrict__ h16_0,
                               float* __restrict__ c) {
    int i = blockIdx.x * blockDim.x + threadIdx.x;     // 0 .. B*H-1 (131072)
    if (i < 4 * H_) {
        int g = i >> 10, n = i & (H_ - 1);
        const float* bi[4] = {bii, bif, big, bio};
        const float* bh[4] = {bhi, bhf, bhg, bho};
        bcomb[i] = bi[g][n] + bh[g][n];
    }
    h16_0[i] = (h16_t)0.f;
    c[i]     = 0.f;
}

// ---------------------------------------------------------------------------
// Fragment loaders (contiguous K per lane half, same order for A and B, wave32)
// ---------------------------------------------------------------------------
// A 16x32 f16: lane = hi*16+r holds row r, K runs [hi*16 .. +15] (contiguous)
__device__ __forceinline__ v16h load_a_f16(const h16_t* p) {
    v16h f;
    ((uint4*)&f)[0] = *(const uint4*)(p);        // 8 halves
    ((uint4*)&f)[1] = *(const uint4*)(p + 8);    // 8 halves
    return f;
}

// B 32x16 f16 from N-major weights: lane = hi*16+r holds column r,
// 16 consecutive K at k0 + hi*16 (one contiguous 32-byte run).
__device__ __forceinline__ v16h load_b(const h16_t* p) {
    return *(const v16h*)p;
}

__device__ __forceinline__ float fsigmoid(float x) { return 1.f / (1.f + __expf(-x)); }
__device__ __forceinline__ float ftanh(float x) {
    float e = __expf(-2.f * x);
    return (1.f - e) / (1.f + e);
}

// ---------------------------------------------------------------------------
// One LSTM step: gates[128,4H] = [x_t | h_in] @ Wt^T + b ; pointwise update.
// grid = (64 n-tiles, 2 m-halves), 128 threads = 4 waves (16 rows each):
// 512 tiles spread across 128 WGPs, 1 wave per SIMD32.
// ---------------------------------------------------------------------------
__global__ void __launch_bounds__(128)
lstm_step(const h16_t* __restrict__ xh, const h16_t* __restrict__ Wt,
          const float* __restrict__ bcomb,
          const h16_t* __restrict__ h_in, h16_t* __restrict__ h_out,
          float* __restrict__ c, float* __restrict__ h32, int t) {
    const int lane = threadIdx.x & 31;
    const int wave = threadIdx.x >> 5;                    // 0..3
    const int m0   = ((blockIdx.y << 2) + wave) << 4;     // batch-row tile (0..112)
    const int n0   = blockIdx.x << 4;                     // gate-local column tile
    const int r    = lane & 15;
    const int hi   = lane >> 4;

    v8f acc0 = {}, acc1 = {}, acc2 = {}, acc3 = {};

    // B-fragment base pointers (per-lane) for the 4 gates.
    const h16_t* b0 = Wt + (size_t)(0 * H_ + n0 + r) * KTOT + hi * 16;
    const h16_t* b1 = Wt + (size_t)(1 * H_ + n0 + r) * KTOT + hi * 16;
    const h16_t* b2 = Wt + (size_t)(2 * H_ + n0 + r) * KTOT + hi * 16;
    const h16_t* b3 = Wt + (size_t)(3 * H_ + n0 + r) * KTOT + hi * 16;

    // Phase 1: K in [0, I) -- x_t rows from pre-converted xh [T,B,I] f16.
    const h16_t* ax = xh + ((size_t)t * B_ + (m0 + r)) * I_ + hi * 16;
#pragma unroll 2
    for (int k = 0; k < I_; k += 32) {
        v16h a  = load_a_f16(ax + k);
        v16h f0 = load_b(b0 + k);
        v16h f1 = load_b(b1 + k);
        v16h f2 = load_b(b2 + k);
        v16h f3 = load_b(b3 + k);
        acc0 = __builtin_amdgcn_wmma_f32_16x16x32_f16(false, a, false, f0, (short)0, acc0, false, false);
        acc1 = __builtin_amdgcn_wmma_f32_16x16x32_f16(false, a, false, f1, (short)0, acc1, false, false);
        acc2 = __builtin_amdgcn_wmma_f32_16x16x32_f16(false, a, false, f2, (short)0, acc2, false, false);
        acc3 = __builtin_amdgcn_wmma_f32_16x16x32_f16(false, a, false, f3, (short)0, acc3, false, false);
    }

    // Phase 2: K in [I, I+H) -- recurrent h (f16 ping-pong buffer).
    const h16_t* ah = h_in + (size_t)(m0 + r) * H_ + hi * 16;
#pragma unroll 2
    for (int k = 0; k < H_; k += 32) {
        v16h a  = load_a_f16(ah + k);
        v16h f0 = load_b(b0 + I_ + k);
        v16h f1 = load_b(b1 + I_ + k);
        v16h f2 = load_b(b2 + I_ + k);
        v16h f3 = load_b(b3 + I_ + k);
        acc0 = __builtin_amdgcn_wmma_f32_16x16x32_f16(false, a, false, f0, (short)0, acc0, false, false);
        acc1 = __builtin_amdgcn_wmma_f32_16x16x32_f16(false, a, false, f1, (short)0, acc1, false, false);
        acc2 = __builtin_amdgcn_wmma_f32_16x16x32_f16(false, a, false, f2, (short)0, acc2, false, false);
        acc3 = __builtin_amdgcn_wmma_f32_16x16x32_f16(false, a, false, f3, (short)0, acc3, false, false);
    }

    // Pointwise gate math. C/D layout: elem v -> row m0+hi*8+v, col n0+(lane&15).
    const int col = n0 + r;
    const float bb0 = bcomb[0 * H_ + col];
    const float bb1 = bcomb[1 * H_ + col];
    const float bb2 = bcomb[2 * H_ + col];
    const float bb3 = bcomb[3 * H_ + col];
#pragma unroll
    for (int v = 0; v < 8; ++v) {
        int row = m0 + hi * 8 + v;
        int idx = row * H_ + col;
        float it = fsigmoid(acc0[v] + bb0);
        float ft = fsigmoid(acc1[v] + bb1);
        float gt = ftanh(acc2[v] + bb2);
        float ot = fsigmoid(acc3[v] + bb3);
        float cn = ft * c[idx] + it * gt;
        float hn = ot * ftanh(cn);
        c[idx]     = cn;
        h32[idx]   = hn;
        h_out[idx] = (h16_t)hn;
    }
}

// ---------------------------------------------------------------------------
extern "C" void kernel_launch(void* const* d_in, const int* in_sizes, int n_in,
                              void* d_out, int out_size, void* d_ws, size_t ws_size,
                              hipStream_t stream) {
    (void)in_sizes; (void)n_in; (void)out_size; (void)ws_size;
    const float* x   = (const float*)d_in[0];
    const float* Wii = (const float*)d_in[1];  const float* Whi = (const float*)d_in[2];
    const float* Wif = (const float*)d_in[3];  const float* Whf = (const float*)d_in[4];
    const float* Wig = (const float*)d_in[5];  const float* Whg = (const float*)d_in[6];
    const float* Wio = (const float*)d_in[7];  const float* Who = (const float*)d_in[8];
    const float* bii = (const float*)d_in[9];  const float* bhi = (const float*)d_in[10];
    const float* bif = (const float*)d_in[11]; const float* bhf = (const float*)d_in[12];
    const float* big = (const float*)d_in[13]; const float* bhg = (const float*)d_in[14];
    const float* bio = (const float*)d_in[15]; const float* bho = (const float*)d_in[16];

    char* ws = (char*)d_ws;
    h16_t* Wt    = (h16_t*)ws;  ws += (size_t)4 * H_ * KTOT * sizeof(h16_t);   // 16 MB
    float* bcomb = (float*)ws;  ws += (size_t)4 * H_ * sizeof(float);          // 16 KB
    h16_t* hA    = (h16_t*)ws;  ws += (size_t)B_ * H_ * sizeof(h16_t);         // 256 KB
    h16_t* hB    = (h16_t*)ws;  ws += (size_t)B_ * H_ * sizeof(h16_t);         // 256 KB
    h16_t* xh    = (h16_t*)ws;  ws += (size_t)B_ * T_ * I_ * sizeof(h16_t);    // 128 MB

    float* h32 = (float*)d_out;             // output h  [B,H]
    float* cst = (float*)d_out + B_ * H_;   // output c  [B,H] (doubles as state)

    pack_weights<<<(4 * H_ * KTOT) / 256, 256, 0, stream>>>(
        Wii, Wif, Wig, Wio, Whi, Whf, Whg, Who, Wt);
    convert_x<<<((size_t)B_ * T_ * I_ / 8) / 256, 256, 0, stream>>>(x, xh);
    prep_bias_init<<<(B_ * H_) / 256, 256, 0, stream>>>(
        bii, bhi, bif, bhf, big, bhg, bio, bho, bcomb, hA, cst);

    dim3 grid(H_ / 16, 2);   // 64 n-tiles x 2 m-halves = 128 blocks
    for (int t = 0; t < T_; ++t) {
        const h16_t* hin  = (t & 1) ? hB : hA;
        h16_t*       hout = (t & 1) ? hA : hB;
        lstm_step<<<grid, 128, 0, stream>>>(xh, Wt, bcomb, hin, hout, cst, h32, t);
    }
}
